// SuperpixelSAT_18829136625865
// MI455X (gfx1250) — matrix operand order repaired and checked by reading, not compiled, not run on hardware
//
#include <hip/hip_runtime.h>
#include <hip/hip_bf16.h>

// ---------------------------------------------------------------------------
// SuperpixelSAT forward for MI455X (gfx1250, wave32, WMMA).
// Dense node linears -> LDS-staged v_wmma_f32_16x16x32_f16 (16-row wave tiles).
// Sparse attention  -> L2-resident atomics (u32 order-preserving max + f32 add).
// ---------------------------------------------------------------------------

typedef __attribute__((ext_vector_type(16))) _Float16 v16h;
typedef __attribute__((ext_vector_type(8)))  _Float16 v8h;
typedef __attribute__((ext_vector_type(8)))  float    v8f;

#define HEADS   2
#define F_HEAD  15
#define F_SIZE  30
#define OUTSZ   10

// ---- order-preserving float <-> u32 (enables native atomicMax on floats) ---
__device__ __forceinline__ unsigned f2ord(float f) {
  unsigned u = __float_as_uint(f);
  return (u & 0x80000000u) ? ~u : (u | 0x80000000u);
}
__device__ __forceinline__ float ord2f(unsigned u) {
  return __uint_as_float((u & 0x80000000u) ? (u ^ 0x80000000u) : ~u);
}

// ---------------------------------------------------------------------------
// WMMA GEMM: Y[n,16] = X[n,fin] @ W[fout,fin]^T + bias[fout]   (fout <= 16)
// Block = 256 threads = 8 waves = 128 rows; one 16-row tile per wave.
// KPAD = fin padded to a multiple of 32 (template -> fully unrolled WMMAs).
//
// LDS staging (zero-padded), then branch-free aligned vector ds_loads:
//   A frag (ISA 7.12.2, 16-bit 16x32): lane l, row M=l%16, half=l/16:
//     halves 0..7  <- sA[row][kb +     half*8 .. +7]   (one 16B ds_load)
//     halves 8..15 <- sA[row][kb + 16 + half*8 .. +7]  (one 16B ds_load)
//   B frag (32x16): lane l, col N=l%16:
//     halves 0..7  <- sW[N][kb +     half*8 .. +7]    (one 16B ds_load)
//     halves 8..15 <- sW[N][kb + 16 + half*8 .. +7]   (one 16B ds_load)
// ---------------------------------------------------------------------------
template <int KPAD>
__global__ __launch_bounds__(256)
void k_gemm16(const float* __restrict__ X, int n, int fin,
              const float* __restrict__ W, int fout,
              const float* __restrict__ bias,
              float* __restrict__ Y)
{
  __shared__ alignas(32) _Float16 sA[128 * KPAD];
  __shared__ alignas(32) _Float16 sW[16 * KPAD];

  const int tid   = threadIdx.x;
  const int lane  = tid & 31;
  const int wave  = tid >> 5;
  const int half  = lane >> 4;           // 0 or 1
  const int mn    = lane & 15;           // A-row / B-col within wave tile
  const int rowB  = blockIdx.x * 128;    // first row of this block

  // ---- cooperative zero-padded staging (coalesced over k) -----------------
  for (int idx = tid; idx < 16 * KPAD; idx += 256) {
    const int r = idx / KPAD, k = idx - r * KPAD;
    sW[idx] = (_Float16)((r < fout && k < fin) ? W[r * fin + k] : 0.0f);
  }
  for (int idx = tid; idx < 128 * KPAD; idx += 256) {
    const int r = idx / KPAD, k = idx - r * KPAD;
    const int g = rowB + r;
    sA[idx] = (_Float16)((g < n && k < fin) ? X[(long)g * fin + k] : 0.0f);
  }
  __syncthreads();

  const int row0 = rowB + wave * 16;     // wave-uniform tile base
  if (row0 < n) {                        // EXEC all-ones inside
    const int rl = wave * 16 + mn;       // this lane's A row in LDS
    v8f c = {};
#pragma unroll
    for (int kb = 0; kb < KPAD; kb += 32) {
      const v8h a0 = *(const v8h*)&sA[rl * KPAD + kb + half * 8];
      const v8h a1 = *(const v8h*)&sA[rl * KPAD + kb + 16 + half * 8];
      const v16h a = __builtin_shufflevector(a0, a1, 0, 1, 2, 3, 4, 5, 6, 7,
                                             8, 9, 10, 11, 12, 13, 14, 15);
      const v8h b0 = *(const v8h*)&sW[mn * KPAD + kb + half * 8];
      const v8h b1 = *(const v8h*)&sW[mn * KPAD + kb + 16 + half * 8];
      const v16h b = __builtin_shufflevector(b0, b1, 0, 1, 2, 3, 4, 5, 6, 7,
                                             8, 9, 10, 11, 12, 13, 14, 15);
      c = __builtin_amdgcn_wmma_f32_16x16x32_f16(false, a, false, b,
                                                 (short)0, c, false, false);
    }
    const float bb = (mn < fout) ? bias[mn] : 0.0f;
#pragma unroll
    for (int r = 0; r < 8; ++r) {
      const int g = row0 + r + half * 8;
      if (g < n) Y[(long)g * 16 + mn] = c[r] + bb;
    }
  }
}

// ---- per-node attention scores: s1 = H[:, :15]@a1w + a1b, s2 likewise -----
__global__ __launch_bounds__(256)
void k_scores(const float* __restrict__ H, int n,
              const float* __restrict__ a1w, const float* __restrict__ a1b,
              const float* __restrict__ a2w, const float* __restrict__ a2b,
              float* __restrict__ s1, float* __restrict__ s2)
{
  const int i = blockIdx.x * blockDim.x + threadIdx.x;
  if (i >= n) return;
  const float* h = H + (long)i * 16;
  float d1 = *a1b, d2 = *a2b;
#pragma unroll
  for (int f = 0; f < F_HEAD; ++f) {
    const float hv = h[f];
    d1 += hv * a1w[f];
    d2 += hv * a2w[f];
  }
  s1[i] = d1;
  s2[i] = d2;
}

// ---- edge pass 1: per-row segment max of leaky_relu(s1[src]+s2[dst]) ------
__global__ __launch_bounds__(256)
void k_edge_max(const int* __restrict__ src, const int* __restrict__ dst,
                int e, const float* __restrict__ s1,
                const float* __restrict__ s2, unsigned* __restrict__ menc)
{
  const int t = blockIdx.x * blockDim.x + threadIdx.x;
  if (t >= e) return;
  const int s = src[t], d = dst[t];
  float v = s1[s] + s2[d];
  v = (v > 0.0f) ? v : 0.01f * v;        // leaky_relu, slope 0.01
  atomicMax(&menc[s], f2ord(v));         // native u32 atomic max in L2
}

// ---- edge pass 2: z[src] += exp(v - m[src]) -------------------------------
__global__ __launch_bounds__(256)
void k_edge_sum(const int* __restrict__ src, const int* __restrict__ dst,
                int e, const float* __restrict__ s1,
                const float* __restrict__ s2,
                const unsigned* __restrict__ menc, float* __restrict__ z)
{
  const int t = blockIdx.x * blockDim.x + threadIdx.x;
  if (t >= e) return;
  const int s = src[t], d = dst[t];
  float v = s1[s] + s2[d];
  v = (v > 0.0f) ? v : 0.01f * v;
  atomicAdd(&z[s], __expf(v - ord2f(menc[s])));
}

// ---- edge pass 3: out[src, colOff:colOff+15] += att * H[dst, :15] ---------
__global__ __launch_bounds__(256)
void k_edge_aggr(const int* __restrict__ src, const int* __restrict__ dst,
                 int e, const float* __restrict__ s1,
                 const float* __restrict__ s2,
                 const unsigned* __restrict__ menc,
                 const float* __restrict__ z,
                 const float* __restrict__ H,
                 float* __restrict__ out, int colOff)
{
  const int t = blockIdx.x * blockDim.x + threadIdx.x;
  if (t >= e) return;
  const int s = src[t], d = dst[t];
  float v = s1[s] + s2[d];
  v = (v > 0.0f) ? v : 0.01f * v;
  const float att = __expf(v - ord2f(menc[s])) / z[s];
  const float4 h0 = *(const float4*)(H + (long)d * 16);
  const float4 h1 = *(const float4*)(H + (long)d * 16 + 4);
  const float4 h2 = *(const float4*)(H + (long)d * 16 + 8);
  const float h12 = H[(long)d * 16 + 12];
  const float h13 = H[(long)d * 16 + 13];
  const float h14 = H[(long)d * 16 + 14];
  float* o = out + (long)s * F_SIZE + colOff;
  atomicAdd(o + 0,  att * h0.x);  atomicAdd(o + 1,  att * h0.y);
  atomicAdd(o + 2,  att * h0.z);  atomicAdd(o + 3,  att * h0.w);
  atomicAdd(o + 4,  att * h1.x);  atomicAdd(o + 5,  att * h1.y);
  atomicAdd(o + 6,  att * h1.z);  atomicAdd(o + 7,  att * h1.w);
  atomicAdd(o + 8,  att * h2.x);  atomicAdd(o + 9,  att * h2.y);
  atomicAdd(o + 10, att * h2.z);  atomicAdd(o + 11, att * h2.w);
  atomicAdd(o + 12, att * h12);   atomicAdd(o + 13, att * h13);
  atomicAdd(o + 14, att * h14);
}

// ---- relu + write next-layer input and skip-concat slice ------------------
__global__ __launch_bounds__(256)
void k_relu_cat(const float* __restrict__ out, int n,
                float* __restrict__ nextin, float* __restrict__ cat, int layer)
{
  const long t = (long)blockIdx.x * blockDim.x + threadIdx.x;
  if (t >= (long)n * F_SIZE) return;
  const float v = fmaxf(out[t], 0.0f);
  nextin[t] = v;
  const long node = t / F_SIZE;
  const int  c    = (int)(t % F_SIZE);
  cat[node * (3 * F_SIZE) + layer * F_SIZE + c] = v;
}

// ---- mean-pool by batch id into comb input x[B,30] ------------------------
__global__ __launch_bounds__(256)
void k_pool(const float* __restrict__ Y, int n, const int* __restrict__ batch,
            float* __restrict__ x, int off, float* __restrict__ cnt)
{
  const int i = blockIdx.x * blockDim.x + threadIdx.x;
  if (i >= n) return;
  const int b = batch[i];
  atomicAdd(&cnt[b], 1.0f);
  const float* y = Y + (long)i * 16;
#pragma unroll
  for (int f = 0; f < OUTSZ; ++f) atomicAdd(&x[b * (3 * OUTSZ) + off + f], y[f]);
}

__global__ __launch_bounds__(256)
void k_pool_div(float* __restrict__ x, int off,
                const float* __restrict__ cnt, int B)
{
  const int b = blockIdx.x * blockDim.x + threadIdx.x;
  if (b >= B) return;
  const float c = fmaxf(cnt[b], 1.0f);
#pragma unroll
  for (int f = 0; f < OUTSZ; ++f) x[b * (3 * OUTSZ) + off + f] /= c;
}

// ---- final classifier + softmax -------------------------------------------
__global__ __launch_bounds__(256)
void k_comb_softmax(const float* __restrict__ x, const float* __restrict__ W,
                    const float* __restrict__ bias, float* __restrict__ outp,
                    int B)
{
  const int row = blockIdx.x * blockDim.x + threadIdx.x;
  if (row >= B) return;
  float logit[OUTSZ];
  float mx = -1e30f;
#pragma unroll
  for (int o = 0; o < OUTSZ; ++o) {
    float acc = bias[o];
    for (int k = 0; k < 3 * OUTSZ; ++k)
      acc += x[row * (3 * OUTSZ) + k] * W[o * (3 * OUTSZ) + k];
    logit[o] = acc;
    mx = fmaxf(mx, acc);
  }
  float zs = 0.0f;
#pragma unroll
  for (int o = 0; o < OUTSZ; ++o) { logit[o] = __expf(logit[o] - mx); zs += logit[o]; }
  const float inv = 1.0f / zs;
#pragma unroll
  for (int o = 0; o < OUTSZ; ++o) outp[row * OUTSZ + o] = logit[o] * inv;
}

// ---------------------------------------------------------------------------
extern "C" void kernel_launch(void* const* d_in, const int* in_sizes, int n_in,
                              void* d_out, int out_size, void* d_ws, size_t ws_size,
                              hipStream_t stream)
{
  (void)n_in; (void)ws_size;
  const int B = out_size / OUTSZ;        // 512

  // ---- inputs -------------------------------------------------------------
  const float* X[3]     = { (const float*)d_in[0], (const float*)d_in[1], (const float*)d_in[2] };
  const int    finX[3]  = { 3, 6, 9 };
  const int    n[3]     = { in_sizes[0] / 3, in_sizes[1] / 6, in_sizes[2] / 9 };
  const int*   batch[3] = { (const int*)d_in[11], (const int*)d_in[12], (const int*)d_in[13] };

  const int ecnt[3] = { in_sizes[3] / 2, in_sizes[5] / 2, in_sizes[9] / 2 };
  const int* eidx[3][2] = {
    { (const int*)d_in[3], (const int*)d_in[3] },   // track0: both heads idx0
    { (const int*)d_in[5], (const int*)d_in[7] },   // track1: idx1u / idx1d
    { (const int*)d_in[9], (const int*)d_in[9] } }; // track2: both heads idx2

  // params flattened in sorted-key pytree order:
  // [14]=comb.W [15]=comb.b, then per track: lin.W, lin.b, 3x(W,a1b,a1w,a2b,a2w,b)
  const float* combW = (const float*)d_in[14];
  const float* combB = (const float*)d_in[15];
  const int tbase[3] = { 16, 36, 56 };

  // ---- workspace layout (floats) ------------------------------------------
  float* ws   = (float*)d_ws;
  float* x    = ws;                        // [B,30] comb input
  float* cnt  = x + (size_t)B * 3 * OUTSZ; // [B]
  float* base = cnt + B;                   // per-track region (tracks serial)

  hipMemsetAsync(x, 0, (size_t)B * 3 * OUTSZ * sizeof(float), stream);

  const int TB = 256;
  for (int tr = 0; tr < 3; ++tr) {
    const int nn = n[tr];
    float*    H      = base;                         // [nn,16] per-head feats
    float*    s1     = H   + (size_t)nn * 16;        // [nn]
    float*    s2     = s1  + nn;                     // [nn]
    unsigned* menc   = (unsigned*)(s2 + nn);         // [nn] encoded seg-max
    float*    z      = (float*)menc + nn;            // [nn]
    float*    out    = z   + nn;                     // [nn,30] layer output
    float*    nextin = out + (size_t)nn * F_SIZE;    // [nn,30] relu'd input
    float*    cat    = nextin + (size_t)nn * F_SIZE; // [nn,90] skip concat

    const float* linW = (const float*)d_in[tbase[tr]];
    const float* linB = (const float*)d_in[tbase[tr] + 1];

    const float* cur = X[tr];
    int fin = finX[tr];

    const int gNode = (nn + TB - 1) / TB;
    const int gTile = (nn + 127) / 128;              // 8 wave-tiles per block
    const int gEl   = (int)(((long)nn * F_SIZE + TB - 1) / TB);
    const int gEdge = (ecnt[tr] + TB - 1) / TB;

    for (int layer = 0; layer < 3; ++layer) {
      const int sb = tbase[tr] + 2 + layer * 6;
      const float* satW = (const float*)d_in[sb];
      const float* a1b  = (const float*)d_in[sb + 1];
      const float* a1w  = (const float*)d_in[sb + 2];
      const float* a2b  = (const float*)d_in[sb + 3];
      const float* a2w  = (const float*)d_in[sb + 4];
      const float* satB = (const float*)d_in[sb + 5];

      hipMemsetAsync(out, 0, (size_t)nn * F_SIZE * sizeof(float), stream);

      for (int head = 0; head < HEADS; ++head) {
        const int* src = eidx[tr][head];
        const int* dst = src + ecnt[tr];

        k_gemm16<32><<<gTile, TB, 0, stream>>>(cur, nn, fin,
                                               satW + head * F_HEAD * fin, F_HEAD,
                                               satB + head * F_HEAD, H);
        k_scores<<<gNode, TB, 0, stream>>>(H, nn,
                                           a1w + head * F_HEAD, a1b + head,
                                           a2w + head * F_HEAD, a2b + head, s1, s2);
        hipMemsetAsync(menc, 0, (size_t)nn * sizeof(unsigned), stream); // 0 < enc(v) for all v
        hipMemsetAsync(z, 0, (size_t)nn * sizeof(float), stream);
        k_edge_max <<<gEdge, TB, 0, stream>>>(src, dst, ecnt[tr], s1, s2, menc);
        k_edge_sum <<<gEdge, TB, 0, stream>>>(src, dst, ecnt[tr], s1, s2, menc, z);
        k_edge_aggr<<<gEdge, TB, 0, stream>>>(src, dst, ecnt[tr], s1, s2, menc, z,
                                              H, out, head * F_HEAD);
      }
      k_relu_cat<<<gEl, TB, 0, stream>>>(out, nn, nextin, cat, layer);
      cur = nextin;
      fin = F_SIZE;
    }

    // skip-concat linear [nn,90] @ [10,90]^T + b (3 unrolled WMMAs, reuse H)
    k_gemm16<96><<<gTile, TB, 0, stream>>>(cat, nn, 3 * F_SIZE, linW, OUTSZ,
                                           linB, H);
    hipMemsetAsync(cnt, 0, (size_t)B * sizeof(float), stream);
    k_pool    <<<gNode, TB, 0, stream>>>(H, nn, batch[tr], x, tr * OUTSZ, cnt);
    k_pool_div<<<(B + TB - 1) / TB, TB, 0, stream>>>(x, tr * OUTSZ, cnt, B);
  }

  k_comb_softmax<<<(B + TB - 1) / TB, TB, 0, stream>>>(x, combW, combB,
                                                       (float*)d_out, B);
}
